// MultiHeadAttentionLayer_4810363372574
// MI455X (gfx1250) — matrix operand order repaired and checked by reading, not compiled, not run on hardware
//
#include <hip/hip_runtime.h>
#include <hip/hip_bf16.h>

// ---------------------------------------------------------------------------
// MI455X (gfx1250, wave32) fused multi-head attention
//   HID=1024, HEADS=16, HEAD_DIM=64, B=4, S=2048, SCALE = 1/8
// Pipeline:
//   cvt(f32->f16) -> proj GEMMs (WMMA f16) -> fused attn (WMMA + softmax)
//   -> output proj GEMM (WMMA f16, f32 epilogue)
// ---------------------------------------------------------------------------

typedef __attribute__((ext_vector_type(16))) _Float16 v16h;
typedef __attribute__((ext_vector_type(8)))  _Float16 v8h;
typedef __attribute__((ext_vector_type(8)))  float    v8f;

#define WMMA_F16(A, B, C) \
  __builtin_amdgcn_wmma_f32_16x16x32_f16(false, (A), false, (B), (short)0, (C), false, false)

#define S_LEN   2048
#define HEADS   16
#define HDIM    64
#define HID     1024
#define SROW    2052   // padded LDS row stride (f32) -> conflict-free g0/g1 halves

// ---- fragment loaders (16-bit A 16x32 / B 32x16 layouts per CDNA5 ISA) ----

// A-matrix 16x32: lane (g = lane/16, m = lane%16), element e:
//   K = (e/8)*16 + g*8 + (e%8)   -> two contiguous 8-half chunks per lane
__device__ __forceinline__ v16h load_a16(const _Float16* __restrict__ base,
                                         int ld, int row0, int k0) {
  const int lane = threadIdx.x & 31;
  const int g = lane >> 4, m = lane & 15;
  const _Float16* p = base + (size_t)(row0 + m) * ld + k0 + g * 8;
  v8h lo = *(const v8h*)(p);
  v8h hi = *(const v8h*)(p + 16);
  v16h a;
#pragma unroll
  for (int i = 0; i < 8; ++i) { a[i] = lo[i]; a[i + 8] = hi[i]; }
  return a;
}

// B-matrix 32x16 built from a ROW-major matrix M[col][k] (i.e. B = M^T):
// lane (g, n = lane%16), element e: K = (e/8)*16 + g*8 + (e%8)  -> two contiguous 8-half chunks
__device__ __forceinline__ v16h load_b16(const _Float16* __restrict__ base,
                                         int ld, int col0, int k0) {
  const int lane = threadIdx.x & 31;
  const int g = lane >> 4, n = lane & 15;
  const _Float16* p = base + (size_t)(col0 + n) * ld + k0 + g * 8;
  v8h lo = *(const v8h*)(p);
  v8h hi = *(const v8h*)(p + 16);
  v16h b;
#pragma unroll
  for (int i = 0; i < 8; ++i) { b[i] = lo[i]; b[i + 8] = hi[i]; }
  return b;
}

// ---------------------------------------------------------------------------
__global__ void cvt_f32_f16(const float* __restrict__ in,
                            _Float16* __restrict__ out, int n) {
  int i = blockIdx.x * blockDim.x + threadIdx.x;
  int stride = gridDim.x * blockDim.x;
  for (; i < n; i += stride) out[i] = (_Float16)in[i];
}

// ---------------------------------------------------------------------------
// Projection: out = X @ W^T + bias,  X:[8192,1024] f16, W:[1024,1024] f16.
// Block = 256 thr (8 waves); wave tile 32x64; block tile 32x512.
// vTranspose==0: write head-split [B,H,S,64]; ==1: write V^T [B,H,64,S].
__global__ void proj_kernel(const _Float16* __restrict__ X,
                            const _Float16* __restrict__ W,
                            const float* __restrict__ bias,
                            _Float16* __restrict__ out, int vTranspose) {
  const int lane = threadIdx.x & 31;
  const int wave = threadIdx.x >> 5;
  const int g = lane >> 4, n = lane & 15;
  const int m0 = blockIdx.y * 32;
  const int n0 = blockIdx.x * 512 + wave * 64;

  v8f acc[2][4] = {};
  for (int k = 0; k < HID; k += 32) {
    v16h a0 = load_a16(X, HID, m0,      k);
    v16h a1 = load_a16(X, HID, m0 + 16, k);
    v16h b0 = load_b16(W, HID, n0,      k);
    v16h b1 = load_b16(W, HID, n0 + 16, k);
    v16h b2 = load_b16(W, HID, n0 + 32, k);
    v16h b3 = load_b16(W, HID, n0 + 48, k);
    acc[0][0] = WMMA_F16(a0, b0, acc[0][0]);
    acc[0][1] = WMMA_F16(a0, b1, acc[0][1]);
    acc[0][2] = WMMA_F16(a0, b2, acc[0][2]);
    acc[0][3] = WMMA_F16(a0, b3, acc[0][3]);
    acc[1][0] = WMMA_F16(a1, b0, acc[1][0]);
    acc[1][1] = WMMA_F16(a1, b1, acc[1][1]);
    acc[1][2] = WMMA_F16(a1, b2, acc[1][2]);
    acc[1][3] = WMMA_F16(a1, b3, acc[1][3]);
  }
#pragma unroll
  for (int i = 0; i < 2; ++i)
#pragma unroll
    for (int j = 0; j < 4; ++j) {
      const int col = n0 + 16 * j + n;
      const float bv = bias[col];
      const int hh = col >> 6, d = col & 63;
#pragma unroll
      for (int r = 0; r < 8; ++r) {
        const int row = m0 + 16 * i + r + 8 * g;       // C: M = r + 8*g
        const int bb = row >> 11, s = row & (S_LEN - 1);
        const _Float16 hv = (_Float16)(acc[i][j][r] + bv);
        const size_t idx = vTranspose
            ? ((size_t)(bb * HEADS + hh) * HDIM + d) * S_LEN + s
            : ((size_t)(bb * HEADS + hh) * S_LEN + s) * HDIM + d;
        out[idx] = hv;
      }
    }
}

// ---------------------------------------------------------------------------
// Output projection: d_out.x = Xh @ Wo^T + bo   (f32 epilogue, [B,S,HID])
__global__ void proj_out_kernel(const _Float16* __restrict__ X,
                                const _Float16* __restrict__ W,
                                const float* __restrict__ bias,
                                float* __restrict__ out) {
  const int lane = threadIdx.x & 31;
  const int wave = threadIdx.x >> 5;
  const int g = lane >> 4, n = lane & 15;
  const int m0 = blockIdx.y * 32;
  const int n0 = blockIdx.x * 512 + wave * 64;

  v8f acc[2][4] = {};
  for (int k = 0; k < HID; k += 32) {
    v16h a0 = load_a16(X, HID, m0,      k);
    v16h a1 = load_a16(X, HID, m0 + 16, k);
    v16h b0 = load_b16(W, HID, n0,      k);
    v16h b1 = load_b16(W, HID, n0 + 16, k);
    v16h b2 = load_b16(W, HID, n0 + 32, k);
    v16h b3 = load_b16(W, HID, n0 + 48, k);
    acc[0][0] = WMMA_F16(a0, b0, acc[0][0]);
    acc[0][1] = WMMA_F16(a0, b1, acc[0][1]);
    acc[0][2] = WMMA_F16(a0, b2, acc[0][2]);
    acc[0][3] = WMMA_F16(a0, b3, acc[0][3]);
    acc[1][0] = WMMA_F16(a1, b0, acc[1][0]);
    acc[1][1] = WMMA_F16(a1, b1, acc[1][1]);
    acc[1][2] = WMMA_F16(a1, b2, acc[1][2]);
    acc[1][3] = WMMA_F16(a1, b3, acc[1][3]);
  }
#pragma unroll
  for (int i = 0; i < 2; ++i)
#pragma unroll
    for (int j = 0; j < 4; ++j) {
      const int col = n0 + 16 * j + n;
      const float bv = bias[col];
#pragma unroll
      for (int r = 0; r < 8; ++r) {
        const int row = m0 + 16 * i + r + 8 * g;
        out[(size_t)row * HID + col] = acc[i][j][r] + bv;
      }
    }
}

// ---------------------------------------------------------------------------
// Fused attention: one workgroup (256 thr / 8 waves) per (b, h, 16-query tile)
//   Phase 1: energy tile 16xS -> LDS (waves split 128 key tiles; 2 WMMA each)
//   Phase 2: rowwise softmax (shfl reductions), write P once to d_out.attn
//   Phase 3: x = P @ V via WMMA (waves split K), LDS cross-wave reduce
__global__ void attn_kernel(const _Float16* __restrict__ Q,
                            const _Float16* __restrict__ K,
                            const _Float16* __restrict__ Vt,
                            float* __restrict__ attnOut,
                            _Float16* __restrict__ Xh) {
  extern __shared__ float sc[];                 // 16 * SROW f32  (~128 KB)
  const int lane = threadIdx.x & 31;
  const int wave = threadIdx.x >> 5;
  const int g = lane >> 4, n = lane & 15;
  const int bid = blockIdx.x;
  const int qt = bid & 127;                     // 128 query tiles
  const int bh = bid >> 7;                      // b*16 + h, 0..63
  const int q0 = qt * 16;

  const _Float16* Qb = Q  + (size_t)bh * S_LEN * HDIM;
  const _Float16* Kb = K  + (size_t)bh * S_LEN * HDIM;
  const _Float16* Vb = Vt + (size_t)bh * HDIM * S_LEN;   // [64][2048]
  float* attnBase = attnOut + ((size_t)bh * S_LEN + q0) * S_LEN;

  // ---- Phase 1: scaled energy into LDS ----
  v16h qa0 = load_a16(Qb, HDIM, q0, 0);
  v16h qa1 = load_a16(Qb, HDIM, q0, 32);
  for (int kt = wave; kt < 128; kt += 8) {
    v16h kb0 = load_b16(Kb, HDIM, kt * 16, 0);
    v16h kb1 = load_b16(Kb, HDIM, kt * 16, 32);
    v8f e = {};
    e = WMMA_F16(qa0, kb0, e);
    e = WMMA_F16(qa1, kb1, e);
#pragma unroll
    for (int r = 0; r < 8; ++r)
      sc[(r + 8 * g) * SROW + kt * 16 + n] = e[r] * 0.125f;
  }
  __syncthreads();

  // ---- Phase 2: softmax (wave handles rows 2w, 2w+1), write P to global ----
#pragma unroll
  for (int rr = 0; rr < 2; ++rr) {
    const int row = wave * 2 + rr;
    float* srow = sc + row * SROW;
    float mx = -3.0e38f;
    for (int c = lane; c < S_LEN; c += 32) mx = fmaxf(mx, srow[c]);
#pragma unroll
    for (int off = 16; off > 0; off >>= 1) mx = fmaxf(mx, __shfl_xor(mx, off, 32));
    float sum = 0.f;
    for (int c = lane; c < S_LEN; c += 32) {
      float ev = __expf(srow[c] - mx);
      srow[c] = ev;
      sum += ev;
    }
#pragma unroll
    for (int off = 16; off > 0; off >>= 1) sum += __shfl_xor(sum, off, 32);
    const float inv = 1.f / sum;
    float* grow = attnBase + (size_t)row * S_LEN;
    for (int c = lane; c < S_LEN; c += 32) {
      float p = srow[c] * inv;
      srow[c] = p;            // keep normalized P in LDS for phase 3
      grow[c] = p;            // the one mandatory 1 GiB write
    }
  }
  __syncthreads();

  // ---- Phase 3: x-tile (16x64) = P @ V ; wave w covers K in [w*256, w*256+256) ----
  v8f xa[4] = {};
  const int m = lane & 15;
  for (int step = 0; step < 8; ++step) {
    const int k0 = wave * 256 + step * 32;
    v16h pa;                                     // A-frag from LDS, f32->f16
    const float* p0 = sc + m * SROW + k0 + g * 8;
#pragma unroll
    for (int i = 0; i < 8; ++i) {
      pa[i]     = (_Float16)p0[i];
      pa[i + 8] = (_Float16)p0[16 + i];
    }
#pragma unroll
    for (int j = 0; j < 4; ++j) {
      v16h vb = load_b16(Vb, S_LEN, j * 16, k0); // contiguous (V stored transposed)
      xa[j] = WMMA_F16(pa, vb, xa[j]);
    }
  }
  __syncthreads();                               // done reading P
  float* part = sc;                              // reuse: [8 waves][16*64] f32
#pragma unroll
  for (int j = 0; j < 4; ++j)
#pragma unroll
    for (int r = 0; r < 8; ++r)
      part[wave * 1024 + (r + 8 * g) * HDIM + j * 16 + n] = xa[j][r];
  __syncthreads();
  const int b = bh >> 4, h = bh & 15;
  for (int idx = threadIdx.x; idx < 1024; idx += 256) {
    float s = 0.f;
#pragma unroll
    for (int ww = 0; ww < 8; ++ww) s += part[ww * 1024 + idx];
    const int row = idx >> 6, d = idx & 63;
    Xh[((size_t)b * S_LEN + q0 + row) * HID + h * HDIM + d] = (_Float16)s;
  }
}

// ---------------------------------------------------------------------------
extern "C" void kernel_launch(void* const* d_in, const int* in_sizes, int n_in,
                              void* d_out, int out_size, void* d_ws, size_t ws_size,
                              hipStream_t stream) {
  (void)in_sizes; (void)n_in; (void)out_size; (void)ws_size;
  const float* q  = (const float*)d_in[0];
  const float* k  = (const float*)d_in[1];
  const float* v  = (const float*)d_in[2];
  const float* Wq = (const float*)d_in[3];
  const float* bq = (const float*)d_in[4];
  const float* Wk = (const float*)d_in[5];
  const float* bk = (const float*)d_in[6];
  const float* Wv = (const float*)d_in[7];
  const float* bv = (const float*)d_in[8];
  const float* Wo = (const float*)d_in[9];
  const float* bo = (const float*)d_in[10];

  char* ws = (char*)d_ws;
  const size_t MB = 1024 * 1024;
  _Float16* qh  = (_Float16*)(ws +   0 * MB);   // 16 MiB each
  _Float16* kh  = (_Float16*)(ws +  16 * MB);
  _Float16* vh  = (_Float16*)(ws +  32 * MB);
  _Float16* Wqh = (_Float16*)(ws +  48 * MB);   // 2 MiB each
  _Float16* Wkh = (_Float16*)(ws +  50 * MB);
  _Float16* Wvh = (_Float16*)(ws +  52 * MB);
  _Float16* Woh = (_Float16*)(ws +  54 * MB);
  _Float16* Qh  = (_Float16*)(ws +  56 * MB);   // [B,H,S,64] f16
  _Float16* Kh  = (_Float16*)(ws +  72 * MB);   // [B,H,S,64] f16
  _Float16* Vth = (_Float16*)(ws +  88 * MB);   // [B,H,64,S] f16 (transposed)
  _Float16* Xh  = (_Float16*)(ws + 104 * MB);   // [B,S,HID]  f16  (ends 120 MiB)

  const int NACT = 4 * S_LEN * HID;             // 8,388,608
  const int NW   = HID * HID;                   // 1,048,576

  cvt_f32_f16<<<2048, 256, 0, stream>>>(q,  qh,  NACT);
  cvt_f32_f16<<<2048, 256, 0, stream>>>(k,  kh,  NACT);
  cvt_f32_f16<<<2048, 256, 0, stream>>>(v,  vh,  NACT);
  cvt_f32_f16<<< 512, 256, 0, stream>>>(Wq, Wqh, NW);
  cvt_f32_f16<<< 512, 256, 0, stream>>>(Wk, Wkh, NW);
  cvt_f32_f16<<< 512, 256, 0, stream>>>(Wv, Wvh, NW);
  cvt_f32_f16<<< 512, 256, 0, stream>>>(Wo, Woh, NW);

  dim3 pgrid(2, 256);                           // N/512 x M/32
  proj_kernel<<<pgrid, 256, 0, stream>>>(qh, Wqh, bq, Qh, 0);
  proj_kernel<<<pgrid, 256, 0, stream>>>(kh, Wkh, bk, Kh, 0);
  proj_kernel<<<pgrid, 256, 0, stream>>>(vh, Wvh, bv, Vth, 1);

  float* out_x    = (float*)d_out;              // [4,2048,1024]
  float* out_attn = out_x + NACT;               // [4,16,2048,2048]
  const size_t smem = (size_t)16 * SROW * sizeof(float);  // ~128.3 KB of 320 KB WGP LDS
  attn_kernel<<<4 * HEADS * (S_LEN / 16), 256, smem, stream>>>(Qh, Kh, Vth, out_attn, Xh);

  proj_out_kernel<<<pgrid, 256, 0, stream>>>(Xh, Woh, bo, out_x);
}
